// TensorEmbedding_71313636983495
// MI455X (gfx1250) — matrix operand order repaired and checked by reading, not compiled, not run on hardware
//
#include <hip/hip_runtime.h>
#include <hip/hip_bf16.h>
#include <math.h>

typedef __attribute__((ext_vector_type(16))) _Float16 v16h;
typedef __attribute__((ext_vector_type(8)))  float    v8f;

#define HCH 64

__device__ __forceinline__ float sigf(float x){ return 1.0f/(1.0f+__expf(-x)); }

// ---------------- K0: zero the scatter accumulator ----------------
__global__ void k_zero(float* p, int n){
  int i = blockIdx.x*blockDim.x + threadIdx.x;
  int stride = gridDim.x*blockDim.x;
  for (; i < n; i += stride) p[i] = 0.0f;
}

// ---------------- Kp: pack weights into WMMA B-fragment layout (f16) ----------
// packDP : 12 frags (grp*4+ht)   : B[k][n] = dpW[grp][ht*16+n][k]
// packUV : 32 frags (out*16+t*4+kk): B[k][n] = WU/WV[t*16+n][kk*32..]
// fragment storage: frag*512 + lane*16 + hh   (lane n = lane&15, k = 8*(lane>>4) + (hh<8 ? hh : 8+hh))
__global__ void k_pack(const float* __restrict__ emb2_W, const float* __restrict__ lemb2_W,
                       const float* __restrict__ dp1W, const float* __restrict__ dp2W,
                       const float* __restrict__ dp3W,
                       _Float16* __restrict__ packDP, _Float16* __restrict__ packUV){
  const float* dpW[3] = {dp1W, dp2W, dp3W};
  int i = blockIdx.x*blockDim.x + threadIdx.x;
  int stride = gridDim.x*blockDim.x;
  for (int idx = i; idx < 12*512; idx += stride){
    int frag = idx >> 9, rem = idx & 511;
    int lane = rem >> 4, hh = rem & 15;
    int grp = frag >> 2, ht = frag & 3;
    int h = ht*16 + (lane & 15);
    int k = ((lane < 16) ? 0 : 8) + ((hh < 8) ? hh : 8 + hh);
    packDP[idx] = (_Float16)dpW[grp][h*32 + k];
  }
  for (int idx = i; idx < 32*512; idx += stride){
    int frag = idx >> 9, rem = idx & 511;
    int lane = rem >> 4, hh = rem & 15;
    int out = frag >> 4;            // 0 = U (src cols), 1 = V (dst cols)
    int t = (frag >> 2) & 3, kk = frag & 3;
    int g = t*16 + (lane & 15);
    int kglob = kk*32 + ((lane < 16) ? 0 : 8) + ((hh < 8) ? hh : 8 + hh);
    float val;
    if (out == 0)
      val = (kglob < 64) ? emb2_W[(size_t)g*128 + kglob]
                         : lemb2_W[(size_t)g*128 + (kglob-64)];
    else
      val = (kglob < 64) ? emb2_W[(size_t)g*128 + 64 + kglob]
                         : lemb2_W[(size_t)g*128 + 64 + (kglob-64)];
    packUV[idx] = (_Float16)val;
  }
}

// ---------------- K1: per-node U,V  (WMMA 16x16x32 f16, 4 tiles/block) --------
__global__ void __launch_bounds__(128)
k_node_uv(const int* __restrict__ z, const float* __restrict__ labels,
          const unsigned char* __restrict__ lmask,
          const float* __restrict__ emb_W,
          const float* __restrict__ lab_W, const float* __restrict__ lab_b,
          const float* __restrict__ mask_tok, const _Float16* __restrict__ packUV,
          float* __restrict__ U, float* __restrict__ V, int N, int ntiles){
  __shared__ _Float16 sX[4][16][128];
  const int lane = threadIdx.x & 31;
  const int w    = threadIdx.x >> 5;
  const int tile = blockIdx.x*4 + w;
  const bool active = tile < ntiles;
  const int n0 = tile*16;
  if (active){
    int r = lane & 15;
    int n = n0 + r; if (n >= N) n = N-1;
    if (lane < 16) {
      const float* zr = emb_W + (size_t)z[n]*HCH;
      #pragma unroll 8
      for (int k=0;k<HCH;k++) sX[w][r][k] = (_Float16)zr[k];
    } else {
      float lv = labels[n];
      bool m = lmask[n] != 0;
      #pragma unroll 8
      for (int k=0;k<HCH;k++){
        float val = m ? mask_tok[k] : (lv*lab_W[k] + lab_b[k]);
        sX[w][r][HCH+k] = (_Float16)val;
      }
    }
  }
  __syncthreads();
  if (!active) return;
  const int Mrow = lane & 15;
  const int hi   = (lane < 16) ? 0 : 1;
  // A-fragments for all 4 K-steps (ISA 16-bit A layout)
  v16h a[4];
  #pragma unroll
  for (int kk=0; kk<4; kk++){
    const int base = hi ? 8 : 0;
    #pragma unroll
    for (int h2=0;h2<8;h2++) a[kk][h2]   = sX[w][Mrow][kk*32 + base + h2];
    #pragma unroll
    for (int h2=0;h2<8;h2++) a[kk][8+h2] = sX[w][Mrow][kk*32 + 16 + base + h2];
  }
  const v16h* pUV = (const v16h*)packUV;
  #pragma unroll
  for (int t=0;t<4;t++){
    v8f accU = {}, accV = {};
    #pragma unroll
    for (int kk=0; kk<4; kk++){
      v16h bu = pUV[(t*4+kk)*32 + lane];
      v16h bv = pUV[(16 + t*4+kk)*32 + lane];
      accU = __builtin_amdgcn_wmma_f32_16x16x32_f16(false, a[kk], false, bu, (short)0, accU, false, false);
      accV = __builtin_amdgcn_wmma_f32_16x16x32_f16(false, a[kk], false, bv, (short)0, accV, false, false);
    }
    #pragma unroll
    for (int r=0;r<8;r++){
      int M = r + (hi?8:0);
      int n = n0 + M;
      if (n < N) {
        U[(size_t)n*HCH + t*16 + Mrow] = accU[r];
        V[(size_t)n*HCH + t*16 + Mrow] = accV[r];
      }
    }
  }
}

// ---------------- K2: edge GEMM (WMMA) + gated moment scatter ----------------
__global__ void __launch_bounds__(256)
k_edge(const int* __restrict__ ei, const float* __restrict__ ew,
       const float* __restrict__ evec, const float* __restrict__ attr,
       const float* __restrict__ emb2_b, const float* __restrict__ lemb2_b,
       const _Float16* __restrict__ packDP,
       const float* __restrict__ dp1b, const float* __restrict__ dp2b,
       const float* __restrict__ dp3b,
       const float* __restrict__ Uu, const float* __restrict__ Vv,
       float* __restrict__ acc, int E, int ntiles){
  __shared__ int   s_src[8][16];
  __shared__ int   s_dst[8][16];
  __shared__ float s_cut[8][16];
  __shared__ float s_G[8][16][9];
  const int lane = threadIdx.x & 31;
  const int w    = threadIdx.x >> 5;
  const int tile = blockIdx.x*8 + w;
  const bool active = tile < ntiles;
  const int e0 = tile*16;
  if (active && lane < 16){
    int e = e0 + lane;
    float cut=0.0f, vx=0.0f, vy=0.0f, vz=0.0f; int s=0, d=0;
    if (e < E){
      s = ei[e]; d = ei[E+e];
      float wgt = ew[e];
      cut = (wgt < 5.0f) ? 0.5f*(__cosf(wgt*0.62831853071795864769f)+1.0f) : 0.0f;
      vx = evec[(size_t)e*3+0]; vy = evec[(size_t)e*3+1]; vz = evec[(size_t)e*3+2];
    }
    // warm L2/L0 for the only random-access gathers of this kernel
    __builtin_prefetch(&Uu[(size_t)s*HCH], 0, 0);
    __builtin_prefetch(&Uu[(size_t)s*HCH + 32], 0, 0);
    __builtin_prefetch(&Vv[(size_t)d*HCH], 0, 0);
    __builtin_prefetch(&Vv[(size_t)d*HCH + 32], 0, 0);
    s_src[w][lane]=s; s_dst[w][lane]=d; s_cut[w][lane]=cut;
    s_G[w][lane][0]=vx;    s_G[w][lane][1]=vy;    s_G[w][lane][2]=vz;
    s_G[w][lane][3]=vx*vx; s_G[w][lane][4]=vy*vy; s_G[w][lane][5]=vz*vz;
    s_G[w][lane][6]=vx*vy; s_G[w][lane][7]=vx*vz; s_G[w][lane][8]=vy*vz;
  }
  __syncthreads();
  if (!active) return;
  const int Mrow = lane & 15;
  const int hi   = (lane<16)?0:1;
  // A-fragment: 16 edges x K=32 (edge_attr), ISA 16-bit A layout
  int ea = e0 + Mrow; if (ea >= E) ea = E-1;
  const float* arow = attr + (size_t)ea*32;
  v16h a;
  { const int base = hi?8:0;
    #pragma unroll
    for (int h2=0;h2<8;h2++) a[h2]   = (_Float16)arow[base+h2];
    #pragma unroll
    for (int h2=0;h2<8;h2++) a[8+h2] = (_Float16)arow[16+base+h2];
  }
  const v16h* pDP = (const v16h*)packDP;
  const float* dpb[3] = {dp1b, dp2b, dp3b};
  for (int ht=0; ht<4; ht++){
    const int hch = ht*16 + Mrow;               // channel for this lane's column
    const float bs = emb2_b[hch] + lemb2_b[hch];
    float wc[8]; int nodes[8];
    #pragma unroll
    for (int r=0;r<8;r++){
      int M = r + (hi?8:0);
      int sN = s_src[w][M], dN = s_dst[w][M];
      nodes[r] = sN;
      wc[r] = s_cut[w][M]*(Uu[(size_t)sN*HCH+hch] + Vv[(size_t)dN*HCH+hch] + bs);
    }
    #pragma unroll
    for (int grp=0; grp<3; grp++){
      v16h b = pDP[(grp*4+ht)*32 + lane];
      v8f c = {};
      c = __builtin_amdgcn_wmma_f32_16x16x32_f16(false, a, false, b, (short)0, c, false, false);
      const float bias = dpb[grp][hch];
      #pragma unroll
      for (int r=0;r<8;r++){
        int M = r + (hi?8:0);
        float f = (c[r] + bias) * wc[r];
        float* p = acc + ((size_t)nodes[r]*HCH + hch)*10;
        if (grp==0){
          atomicAdd(p+0, f);
        } else if (grp==1){
          atomicAdd(p+1, f*s_G[w][M][0]);
          atomicAdd(p+2, f*s_G[w][M][1]);
          atomicAdd(p+3, f*s_G[w][M][2]);
        } else {
          atomicAdd(p+4, f*s_G[w][M][3]);
          atomicAdd(p+5, f*s_G[w][M][4]);
          atomicAdd(p+6, f*s_G[w][M][5]);
          atomicAdd(p+7, f*s_G[w][M][6]);
          atomicAdd(p+8, f*s_G[w][M][7]);
          atomicAdd(p+9, f*s_G[w][M][8]);
        }
      }
    }
  }
}

// ---------------- K3: tensor-norm -> LN -> MLP gates -> channel mix ----------
__global__ void __launch_bounds__(64)
k_node_out(const float* __restrict__ acc,
           const float* __restrict__ ln_g, const float* __restrict__ ln_b,
           const float* __restrict__ ls0W, const float* __restrict__ ls0b,
           const float* __restrict__ ls1W, const float* __restrict__ ls1b,
           const float* __restrict__ lt0W, const float* __restrict__ lt1W,
           const float* __restrict__ lt2W, float* __restrict__ out, int N){
  __shared__ float sA[64][10];
  __shared__ float sTn[64];
  __shared__ float sNrm[64];
  __shared__ float sH1[128];
  __shared__ float sGate[192];
  const int t = threadIdx.x;
  const int n = blockIdx.x;
  const float* ap = acc + ((size_t)n*HCH + t)*10;
  float c0=ap[0], ax=ap[1], ay=ap[2], az=ap[3];
  float xx=ap[4], yy=ap[5], zz=ap[6], xy=ap[7], xz=ap[8], yz=ap[9];
  #pragma unroll
  for (int c=0;c<10;c++) sA[t][c] = ap[c];
  float tr3 = (xx+yy+zz)*(1.0f/3.0f);
  float dx=c0+xx-tr3, dy=c0+yy-tr3, dz=c0+zz-tr3;
  // (u+v)^2 + (u-v)^2 = 2u^2 + 2v^2 for the off-diagonal pairs
  float tn = dx*dx+dy*dy+dz*dz + 2.0f*((xy*xy+az*az)+(xz*xz+ay*ay)+(yz*yz+ax*ax));
  sTn[t] = tn;
  __syncthreads();
  float s=0.0f, s2=0.0f;
  #pragma unroll 8
  for (int k=0;k<64;k++){ float v=sTn[k]; s+=v; s2+=v*v; }
  float mu  = s*(1.0f/64.0f);
  float var = s2*(1.0f/64.0f) - mu*mu;
  sNrm[t] = (tn-mu)*rsqrtf(var+1e-5f)*ln_g[t] + ln_b[t];
  __syncthreads();
  for (int o=t; o<128; o+=64){
    float av = ls0b[o];
    const float* wrow = ls0W + (size_t)o*64;
    #pragma unroll 8
    for (int k=0;k<64;k++) av += wrow[k]*sNrm[k];
    sH1[o] = av*sigf(av);
  }
  __syncthreads();
  #pragma unroll
  for (int c=0;c<3;c++){
    int o = t*3+c;
    float av = ls1b[o];
    const float* wrow = ls1W + (size_t)o*128;
    #pragma unroll 8
    for (int k=0;k<128;k++) av += wrow[k]*sH1[k];
    sGate[o] = av*sigf(av);
  }
  __syncthreads();
  float m[10];
  #pragma unroll
  for (int c=0;c<10;c++) m[c]=0.0f;
  const float* w0 = lt0W + (size_t)t*64;
  const float* w1 = lt1W + (size_t)t*64;
  const float* w2 = lt2W + (size_t)t*64;
  for (int h2=0; h2<64; h2++){
    float wa=w0[h2], wb=w1[h2], wcc=w2[h2];
    m[0]+=wa*sA[h2][0];
    m[1]+=wb*sA[h2][1];  m[2]+=wb*sA[h2][2];  m[3]+=wb*sA[h2][3];
    m[4]+=wcc*sA[h2][4]; m[5]+=wcc*sA[h2][5]; m[6]+=wcc*sA[h2][6];
    m[7]+=wcc*sA[h2][7]; m[8]+=wcc*sA[h2][8]; m[9]+=wcc*sA[h2][9];
  }
  float g0=sGate[t*3+0], g1=sGate[t*3+1], g2=sGate[t*3+2];
  float trp = (m[4]+m[5]+m[6])*(1.0f/3.0f);
  float* op = out + ((size_t)n*HCH + t)*9;
  op[0] =  m[0]*g0 + (m[4]-trp)*g2;
  op[1] = -m[3]*g1 + m[7]*g2;
  op[2] =  m[2]*g1 + m[8]*g2;
  op[3] =  m[3]*g1 + m[7]*g2;
  op[4] =  m[0]*g0 + (m[5]-trp)*g2;
  op[5] = -m[1]*g1 + m[9]*g2;
  op[6] = -m[2]*g1 + m[8]*g2;
  op[7] =  m[1]*g1 + m[9]*g2;
  op[8] =  m[0]*g0 + (m[6]-trp)*g2;
}

extern "C" void kernel_launch(void* const* d_in, const int* in_sizes, int n_in,
                              void* d_out, int out_size, void* d_ws, size_t ws_size,
                              hipStream_t stream){
  const int*   z      = (const int*)d_in[0];
  const int*   ei     = (const int*)d_in[1];
  const float* ew     = (const float*)d_in[2];
  const float* evec   = (const float*)d_in[3];
  const float* attr   = (const float*)d_in[4];
  const float* labels = (const float*)d_in[5];
  const unsigned char* lmask = (const unsigned char*)d_in[6];
  const float* emb_W  = (const float*)d_in[7];
  const float* emb2_W = (const float*)d_in[8];
  const float* emb2_b = (const float*)d_in[9];
  const float* lab_W  = (const float*)d_in[10];
  const float* lab_b  = (const float*)d_in[11];
  const float* lemb2_W= (const float*)d_in[12];
  const float* lemb2_b= (const float*)d_in[13];
  const float* mtok   = (const float*)d_in[14];
  const float* dp1W   = (const float*)d_in[15];
  const float* dp1b   = (const float*)d_in[16];
  const float* dp2W   = (const float*)d_in[17];
  const float* dp2b   = (const float*)d_in[18];
  const float* dp3W   = (const float*)d_in[19];
  const float* dp3b   = (const float*)d_in[20];
  const float* lt0W   = (const float*)d_in[21];
  const float* lt1W   = (const float*)d_in[22];
  const float* lt2W   = (const float*)d_in[23];
  const float* ls0W   = (const float*)d_in[24];
  const float* ls0b   = (const float*)d_in[25];
  const float* ls1W   = (const float*)d_in[26];
  const float* ls1b   = (const float*)d_in[27];
  const float* ln_g   = (const float*)d_in[28];
  const float* ln_b   = (const float*)d_in[29];
  float* out = (float*)d_out;

  const int N = in_sizes[0];
  const int E = in_sizes[2];

  // scratch: acc [N,64,10] | U [N,64] | V [N,64] | packDP (12*512 h) | packUV (32*512 h)
  float* acc = (float*)d_ws;
  float* U   = acc + (size_t)N*HCH*10;
  float* V   = U   + (size_t)N*HCH;
  _Float16* packDP = (_Float16*)(V + (size_t)N*HCH);
  _Float16* packUV = packDP + 12*512;

  int accN = N*HCH*10;
  int zb = (accN + 255)/256; if (zb > 4096) zb = 4096;
  k_zero<<<zb, 256, 0, stream>>>(acc, accN);

  k_pack<<<64, 256, 0, stream>>>(emb2_W, lemb2_W, dp1W, dp2W, dp3W, packDP, packUV);

  int ntilesN = (N + 15)/16;
  k_node_uv<<<(ntilesN + 3)/4, 128, 0, stream>>>(z, labels, lmask, emb_W,
                                                 lab_W, lab_b, mtok, packUV,
                                                 U, V, N, ntilesN);

  int ntilesE = (E + 15)/16;
  k_edge<<<(ntilesE + 7)/8, 256, 0, stream>>>(ei, ew, evec, attr, emb2_b, lemb2_b,
                                              packDP, dp1b, dp2b, dp3b,
                                              U, V, acc, E, ntilesE);

  k_node_out<<<N, 64, 0, stream>>>(acc, ln_g, ln_b, ls0W, ls0b, ls1W, ls1b,
                                   lt0W, lt1W, lt2W, out, N);
}
